// MPNN_6528350289988
// MI455X (gfx1250) — compile-verified, matrix operand order checked by reading, not run on hardware
//
#include <hip/hip_runtime.h>
#include <hip/hip_bf16.h>
#include <stdint.h>

// ---------------------------------------------------------------------------
// CDNA5 (gfx1250) MPNN: bf16 WMMA 16x16x32, fp32 accumulate.
// One wave32 = TWO 16-row tiles (32 edges/nodes) so each weight (B) fragment
// load feeds two WMMAs. 8 waves / block.
// ---------------------------------------------------------------------------

typedef __attribute__((ext_vector_type(16))) __bf16 v16bf;
typedef __attribute__((ext_vector_type(8)))  float  v8f;

union AF { v16bf v; uint32_t u[8]; };
union CF { v8f   v; float    f[8]; };

#define NWARP 8
#define MT 2            // M-tiles per wave

// K index held in half 0 of dword v for this lane-half (ISA 16-bit A layout):
// lanes 0-15: V0..3 -> K=0..7, V4..7 -> K=16..23 ; lanes 16-31: +8
__device__ __forceinline__ int kmap(int v, int hi) {
  return ((v & 4) << 2) | ((v & 3) << 1) | (hi << 3);
}

__device__ __forceinline__ uint16_t f2bf(float f) {
  uint32_t u = __float_as_uint(f);
  u += 0x7FFFu + ((u >> 16) & 1u);          // round to nearest even
  return (uint16_t)(u >> 16);
}
__device__ __forceinline__ uint32_t pack2(float a, float b) {
  return (uint32_t)f2bf(a) | ((uint32_t)f2bf(b) << 16);
}

__device__ __forceinline__ v8f wmma_bf16(v16bf a, v16bf b, v8f c) {
  return __builtin_amdgcn_wmma_f32_16x16x32_bf16(
      /*neg_a=*/false, a, /*neg_b=*/false, b,
      /*c_mod=*/(short)0, c, /*reuse_a=*/false, /*reuse_b=*/false);
}

// Load a 16x32 fragment row (this lane's row) from contiguous bf16 data.
__device__ __forceinline__ void load_frag_bf(AF& f, const uint16_t* row,
                                             int kbase, int hi) {
#pragma unroll
  for (int v = 0; v < 8; ++v)
    f.u[v] = *(const uint32_t*)(row + kbase + kmap(v, hi));
}
// Same, converting from fp32 source on the fly.
__device__ __forceinline__ void load_frag_f32(AF& f, const float* row,
                                              int kbase, int hi) {
#pragma unroll
  for (int v = 0; v < 8; ++v) {
    int k = kbase + kmap(v, hi);
    float2 t = *(const float2*)(row + k);
    f.u[v] = pack2(t.x, t.y);
  }
}

// ---------------------------------------------------------------------------
// Edge message kernel: m = relu(cat(h[dst],h[src]) @ W1 + b1) @ W2 + b2,
// scatter-add into agg[dst] (fp32 atomics, L2 resident).
// ---------------------------------------------------------------------------
__global__ __launch_bounds__(256) void edge_msg(
    const uint16_t* __restrict__ h,          // N x 64 bf16
    const long long* __restrict__ src,
    const long long* __restrict__ dst,
    const uint16_t* __restrict__ w1t,        // 64 x 128 bf16 (W1^T, n-major)
    const float* __restrict__ b1,
    const uint16_t* __restrict__ w2t,        // 64 x 64 bf16 (W2^T)
    const float* __restrict__ b2,
    float* __restrict__ agg,                 // N x 64 f32
    long long E) {
  __shared__ uint16_t lds[NWARP][MT][16 * 66];  // padded: no bank conflicts
  const int wave = threadIdx.x >> 5, lane = threadIdx.x & 31;
  const int m = lane & 15, hi = lane >> 4;
  long long base = ((long long)blockIdx.x * NWARP + wave) * (16 * MT);
  if (base >= E) return;                     // uniform per wave

  const uint16_t* rowd[MT];
  const uint16_t* rowsrc[MT];
#pragma unroll
  for (int t = 0; t < MT; ++t) {
    long long e = base + t * 16 + m; if (e >= E) e = E - 1;
    rowd[t]   = h + (size_t)(int)dst[e] * 64;   // K 0..63
    rowsrc[t] = h + (size_t)(int)src[e] * 64;   // K 64..127
  }

  CF acc[MT][4];
#pragma unroll
  for (int nt = 0; nt < 4; ++nt) {
    float bv = b1[nt * 16 + m];
#pragma unroll
    for (int t = 0; t < MT; ++t)
#pragma unroll
      for (int r = 0; r < 8; ++r) acc[t][nt].f[r] = bv;
  }
#pragma unroll
  for (int kc = 0; kc < 4; ++kc) {
    AF a[MT];
#pragma unroll
    for (int t = 0; t < MT; ++t)
      load_frag_bf(a[t], (kc < 2) ? rowd[t] : rowsrc[t], (kc & 1) * 32, hi);
#pragma unroll
    for (int nt = 0; nt < 4; ++nt) {
      AF b; load_frag_bf(b, w1t + (size_t)(nt * 16 + m) * 128, kc * 32, hi);
#pragma unroll
      for (int t = 0; t < MT; ++t)
        acc[t][nt].v = wmma_bf16(a[t].v, b.v, acc[t][nt].v);
    }
  }
  // relu + C-layout -> LDS (bf16), then re-read in A-layout
#pragma unroll
  for (int t = 0; t < MT; ++t) {
    uint16_t* sm = lds[wave][t];
#pragma unroll
    for (int nt = 0; nt < 4; ++nt)
#pragma unroll
      for (int r = 0; r < 8; ++r) {
        float vv = acc[t][nt].f[r]; vv = vv > 0.f ? vv : 0.f;
        sm[(r + hi * 8) * 66 + nt * 16 + m] = f2bf(vv);
      }
  }
  asm volatile("s_wait_dscnt 0" ::: "memory");

  CF o[MT][4];
#pragma unroll
  for (int nt = 0; nt < 4; ++nt) {
    float bv = b2[nt * 16 + m];
#pragma unroll
    for (int t = 0; t < MT; ++t)
#pragma unroll
      for (int r = 0; r < 8; ++r) o[t][nt].f[r] = bv;
  }
#pragma unroll
  for (int kc = 0; kc < 2; ++kc) {
    AF a[MT];
#pragma unroll
    for (int t = 0; t < MT; ++t)
      load_frag_bf(a[t], lds[wave][t] + m * 66, kc * 32, hi);
#pragma unroll
    for (int nt = 0; nt < 4; ++nt) {
      AF b; load_frag_bf(b, w2t + (size_t)(nt * 16 + m) * 64, kc * 32, hi);
#pragma unroll
      for (int t = 0; t < MT; ++t)
        o[t][nt].v = wmma_bf16(a[t].v, b.v, o[t][nt].v);
    }
  }
  // scatter-add: this lane owns column (nt*16+m) of rows r+hi*8 of each tile
#pragma unroll
  for (int t = 0; t < MT; ++t) {
    int drow[8];
#pragma unroll
    for (int r = 0; r < 8; ++r) {
      long long er = base + t * 16 + r + hi * 8;
      drow[r] = (er < E) ? (int)dst[er] : -1;
    }
#pragma unroll
    for (int r = 0; r < 8; ++r) {
      if (drow[r] >= 0) {
        float* ap = agg + (size_t)drow[r] * 64 + m;
#pragma unroll
        for (int nt = 0; nt < 4; ++nt) atomicAdd(ap + nt * 16, o[t][nt].f[r]);
      }
    }
  }
}

// ---------------------------------------------------------------------------
// Node MLP kernel: out = relu(in @ W1 + b1) @ W2 + b2
//  emb   : IN1_BF=0 HAS_IN2=0 OUT_F32=0  (x fp32 -> h bf16)
//  update: IN1_BF=1 HAS_IN2=1 OUT_F32=0  (cat(h, agg) -> h, in place)
//  head  : IN1_BF=1 HAS_IN2=0 OUT_F32=1  (h -> d_out fp32)
// ---------------------------------------------------------------------------
template <bool IN1_BF, bool HAS_IN2, bool OUT_F32>
__global__ __launch_bounds__(256) void node_mlp(
    const void* __restrict__ in1, const float* __restrict__ in2,
    const uint16_t* __restrict__ w1t, const float* __restrict__ b1,
    const uint16_t* __restrict__ w2t, const float* __restrict__ b2,
    uint16_t* __restrict__ out_bf, float* __restrict__ out_f, int n) {
  __shared__ uint16_t lds[NWARP][MT][16 * 66];
  const int wave = threadIdx.x >> 5, lane = threadIdx.x & 31;
  const int m = lane & 15, hi = lane >> 4;
  long long base = ((long long)blockIdx.x * NWARP + wave) * (16 * MT);
  if (base >= n) return;
  long long i[MT];
#pragma unroll
  for (int t = 0; t < MT; ++t) {
    i[t] = base + t * 16 + m; if (i[t] >= n) i[t] = n - 1;
  }
  constexpr int KC1 = HAS_IN2 ? 4 : 2;
  constexpr int LDK = KC1 * 32;

  CF acc[MT][4];
#pragma unroll
  for (int nt = 0; nt < 4; ++nt) {
    float bv = b1[nt * 16 + m];
#pragma unroll
    for (int t = 0; t < MT; ++t)
#pragma unroll
      for (int r = 0; r < 8; ++r) acc[t][nt].f[r] = bv;
  }
#pragma unroll
  for (int kc = 0; kc < KC1; ++kc) {
    AF a[MT];
#pragma unroll
    for (int t = 0; t < MT; ++t) {
      if (HAS_IN2 && kc >= 2)
        load_frag_f32(a[t], in2 + i[t] * 64, (kc - 2) * 32, hi);
      else if (IN1_BF)
        load_frag_bf(a[t], (const uint16_t*)in1 + i[t] * 64, (kc & 1) * 32, hi);
      else
        load_frag_f32(a[t], (const float*)in1 + i[t] * 64, (kc & 1) * 32, hi);
    }
#pragma unroll
    for (int nt = 0; nt < 4; ++nt) {
      AF b; load_frag_bf(b, w1t + (size_t)(nt * 16 + m) * LDK, kc * 32, hi);
#pragma unroll
      for (int t = 0; t < MT; ++t)
        acc[t][nt].v = wmma_bf16(a[t].v, b.v, acc[t][nt].v);
    }
  }
#pragma unroll
  for (int t = 0; t < MT; ++t) {
    uint16_t* sm = lds[wave][t];
#pragma unroll
    for (int nt = 0; nt < 4; ++nt)
#pragma unroll
      for (int r = 0; r < 8; ++r) {
        float vv = acc[t][nt].f[r]; vv = vv > 0.f ? vv : 0.f;
        sm[(r + hi * 8) * 66 + nt * 16 + m] = f2bf(vv);
      }
  }
  asm volatile("s_wait_dscnt 0" ::: "memory");

  CF o[MT][4];
#pragma unroll
  for (int nt = 0; nt < 4; ++nt) {
    float bv = b2[nt * 16 + m];
#pragma unroll
    for (int t = 0; t < MT; ++t)
#pragma unroll
      for (int r = 0; r < 8; ++r) o[t][nt].f[r] = bv;
  }
#pragma unroll
  for (int kc = 0; kc < 2; ++kc) {
    AF a[MT];
#pragma unroll
    for (int t = 0; t < MT; ++t)
      load_frag_bf(a[t], lds[wave][t] + m * 66, kc * 32, hi);
#pragma unroll
    for (int nt = 0; nt < 4; ++nt) {
      AF b; load_frag_bf(b, w2t + (size_t)(nt * 16 + m) * 64, kc * 32, hi);
#pragma unroll
      for (int t = 0; t < MT; ++t)
        o[t][nt].v = wmma_bf16(a[t].v, b.v, o[t][nt].v);
    }
  }
#pragma unroll
  for (int t = 0; t < MT; ++t)
#pragma unroll
    for (int r = 0; r < 8; ++r) {
      long long ir = base + t * 16 + r + hi * 8;
      if (ir < n) {
#pragma unroll
        for (int nt = 0; nt < 4; ++nt) {
          if (OUT_F32) out_f[ir * 64 + nt * 16 + m] = o[t][nt].f[r];
          else         out_bf[ir * 64 + nt * 16 + m] = f2bf(o[t][nt].f[r]);
        }
      }
    }
}

// ---------------------------------------------------------------------------
// Weight prep: fp32 W[K][Nc] -> bf16 W^T[Nc][K]
// ---------------------------------------------------------------------------
__global__ void prep_w(const float* __restrict__ w, uint16_t* __restrict__ wt,
                       int K, int Nc) {
  int idx = blockIdx.x * blockDim.x + threadIdx.x;
  if (idx >= K * Nc) return;
  int nn = idx / K, k = idx - nn * K;
  wt[idx] = f2bf(w[(size_t)k * Nc + nn]);
}

__global__ void zero_f(float* __restrict__ p, long long cnt) {
  long long i = (long long)blockIdx.x * blockDim.x + threadIdx.x;
  if (i < cnt) p[i] = 0.f;
}

// ---------------------------------------------------------------------------
extern "C" void kernel_launch(void* const* d_in, const int* in_sizes, int n_in,
                              void* d_out, int out_size, void* d_ws,
                              size_t ws_size, hipStream_t stream) {
  (void)n_in; (void)out_size; (void)ws_size;
  const float* x       = (const float*)d_in[0];
  const long long* ei  = (const long long*)d_in[1];
  const float* emb_w1  = (const float*)d_in[2];
  const float* emb_b1  = (const float*)d_in[3];
  const float* emb_w2  = (const float*)d_in[4];
  const float* emb_b2  = (const float*)d_in[5];
  const float* msg_w1  = (const float*)d_in[6];
  const float* msg_b1  = (const float*)d_in[7];
  const float* msg_w2  = (const float*)d_in[8];
  const float* msg_b2  = (const float*)d_in[9];
  const float* upd_w1  = (const float*)d_in[10];
  const float* upd_b1  = (const float*)d_in[11];
  const float* upd_w2  = (const float*)d_in[12];
  const float* upd_b2  = (const float*)d_in[13];
  const float* head_w1 = (const float*)d_in[14];
  const float* head_b1 = (const float*)d_in[15];
  const float* head_w2 = (const float*)d_in[16];
  const float* head_b2 = (const float*)d_in[17];

  const int L = 4;
  const int N = in_sizes[0] / 64;
  const long long E = in_sizes[1] / 2;
  const long long* srcp = ei;
  const long long* dstp = ei + E;

  char* ws = (char*)d_ws;
  size_t off = 0;
  auto take = [&](size_t b) -> char* {
    char* p = ws + off;
    off = (off + b + 255) & ~(size_t)255;
    return p;
  };
  uint16_t* h   = (uint16_t*)take((size_t)N * 64 * 2);
  float*    agg = (float*)take((size_t)N * 64 * 4);
  uint16_t* ew1 = (uint16_t*)take(64 * 64 * 2);
  uint16_t* ew2 = (uint16_t*)take(64 * 64 * 2);
  uint16_t* mw1 = (uint16_t*)take((size_t)L * 64 * 128 * 2);
  uint16_t* mw2 = (uint16_t*)take((size_t)L * 64 * 64 * 2);
  uint16_t* uw1 = (uint16_t*)take((size_t)L * 64 * 128 * 2);
  uint16_t* uw2 = (uint16_t*)take((size_t)L * 64 * 64 * 2);
  uint16_t* hw1 = (uint16_t*)take(64 * 64 * 2);
  uint16_t* hw2 = (uint16_t*)take(64 * 64 * 2);

  dim3 blk(256);
  auto prep = [&](const float* w, uint16_t* wt, int K, int Nc) {
    int tot = K * Nc;
    prep_w<<<dim3((tot + 255) / 256), blk, 0, stream>>>(w, wt, K, Nc);
  };
  prep(emb_w1, ew1, 64, 64);
  prep(emb_w2, ew2, 64, 64);
  for (int l = 0; l < L; ++l) {
    prep(msg_w1 + (size_t)l * 128 * 64, mw1 + (size_t)l * 64 * 128, 128, 64);
    prep(msg_w2 + (size_t)l * 64 * 64,  mw2 + (size_t)l * 64 * 64,  64, 64);
    prep(upd_w1 + (size_t)l * 128 * 64, uw1 + (size_t)l * 64 * 128, 128, 64);
    prep(upd_w2 + (size_t)l * 64 * 64,  uw2 + (size_t)l * 64 * 64,  64, 64);
  }
  prep(head_w1, hw1, 64, 64);
  prep(head_w2, hw2, 64, 64);

  const int ROWS = 16 * MT;                       // rows per wave
  long long ntiles = ((long long)N + ROWS - 1) / ROWS;
  dim3 ngrid((unsigned)((ntiles + NWARP - 1) / NWARP));
  long long etiles = (E + ROWS - 1) / ROWS;
  dim3 egrid((unsigned)((etiles + NWARP - 1) / NWARP));
  long long aggcnt = (long long)N * 64;
  dim3 zgrid((unsigned)((aggcnt + 255) / 256));

  // embedding: x (fp32) -> h (bf16)
  node_mlp<false, false, false><<<ngrid, blk, 0, stream>>>(
      (const void*)x, (const float*)nullptr, ew1, emb_b1, ew2, emb_b2, h,
      (float*)nullptr, N);

  for (int l = 0; l < L; ++l) {
    zero_f<<<zgrid, blk, 0, stream>>>(agg, aggcnt);
    edge_msg<<<egrid, blk, 0, stream>>>(
        h, srcp, dstp, mw1 + (size_t)l * 64 * 128, msg_b1 + l * 64,
        mw2 + (size_t)l * 64 * 64, msg_b2 + l * 64, agg, E);
    node_mlp<true, true, false><<<ngrid, blk, 0, stream>>>(
        (const void*)h, agg, uw1 + (size_t)l * 64 * 128, upd_b1 + l * 64,
        uw2 + (size_t)l * 64 * 64, upd_b2 + l * 64, h, (float*)nullptr, N);
  }

  // head: h (bf16) -> d_out (fp32)
  node_mlp<true, false, true><<<ngrid, blk, 0, stream>>>(
      (const void*)h, (const float*)nullptr, hw1, head_b1, hw2, head_b2,
      (uint16_t*)nullptr, (float*)d_out, N);
}
